// SpectralTransform_61272003445097
// MI455X (gfx1250) — matrix operand order repaired and checked by reading, not compiled, not run on hardware
//
#include <hip/hip_runtime.h>
#include <hip/hip_fp16.h>

typedef __attribute__((ext_vector_type(16))) _Float16 v16h;
typedef __attribute__((ext_vector_type(8)))  _Float16 v8h;
typedef __attribute__((ext_vector_type(8)))  float    v8f;

#define NPTS    (64*64*64)      // 262144 points per batch
#define NBATCH  2
#define NMODES  128
#define SPLITK  128
#define CHUNK   (NPTS / SPLITK) // 2048 points per K-chunk
#define TILE_P  128             // points staged per stage
#define SUBSTEP (TILE_P / 32)   // 4 WMMA K-steps per stage
#define NSTAGE  (CHUNK / TILE_P)// 16 stages per chunk
#define BROW    20              // padded B-row stride (dwords)
#define OUT_FLOATS (NBATCH * NMODES * NMODES * 2)

__global__ void SpectralTransform_zero(float* __restrict__ out) {
    int i = blockIdx.x * blockDim.x + threadIdx.x;
    if (i < OUT_FLOATS) out[i] = 0.0f;
}

static __device__ __forceinline__ unsigned pack_f16x2(float lo, float hi) {
    unsigned short l = __builtin_bit_cast(unsigned short, (_Float16)lo);
    unsigned short h = __builtin_bit_cast(unsigned short, (_Float16)hi);
    return (unsigned)l | ((unsigned)h << 16);   // v_cvt_pk_f16_f32
}

static __device__ __forceinline__ v16h cat8(v8h lo, v8h hi) {
    return __builtin_shufflevector(lo, hi, 0,1,2,3,4,5,6,7,8,9,10,11,12,13,14,15);
}

__global__ __launch_bounds__(128)
void SpectralTransform_61272003445097_kernel(const float* __restrict__ a,
                                             const float* __restrict__ x,
                                             const float* __restrict__ a_modes,
                                             const float* __restrict__ x_modes,
                                             float* __restrict__ out) {
    __shared__ __align__(64) float    aT[4][TILE_P];           // SoA: a[ch][p]
    __shared__ __align__(64) float4   xS[TILE_P];              // x[p][0..2], pad
    __shared__ __align__(64) unsigned breU[SUBSTEP][32][BROW]; // [sub][s][k/2]
    __shared__ __align__(64) unsigned bimU[SUBSTEP][32][BROW];

    const int chunk = blockIdx.x;
    const int sgrp  = blockIdx.y;        // 32-wide s-group
    const int b     = blockIdx.z;
    const int tid   = threadIdx.x;       // 128 threads = 4 waves
    const int wave  = tid >> 5;
    const int lane  = tid & 31;
    const int half  = lane >> 4;
    const int n     = lane & 15;
    const int m0a   = wave * 16;
    const int m0b   = wave * 16 + 64;
    const int s0    = sgrp * 32;

    // a_modes rows with the 1/sqrt(64^3) output scale pre-folded
    const float scale = 1.0f / 512.0f;
    const float4 amA4 = ((const float4*)a_modes)[m0a + n];
    const float4 amB4 = ((const float4*)a_modes)[m0b + n];
    const float amA[4] = {amA4.x*scale, amA4.y*scale, amA4.z*scale, amA4.w*scale};
    const float amB[4] = {amB4.x*scale, amB4.y*scale, amB4.z*scale, amB4.w*scale};

    // producer: lane = s within 32-wide group; wave = substep (points 32w..32w+31)
    const float NEG2PI = -6.28318530717958647692f;
    const float xm0 = NEG2PI * x_modes[(s0 + lane) * 3 + 0];
    const float xm1 = NEG2PI * x_modes[(s0 + lane) * 3 + 1];
    const float xm2 = NEG2PI * x_modes[(s0 + lane) * 3 + 2];

    v8f accRe[2][2] = {}; // [s-tile][m-tile]
    v8f accIm[2][2] = {};

    const long pbase = (long)b * NPTS + (long)chunk * CHUNK;

    #pragma unroll 1
    for (int t = 0; t < NSTAGE; ++t) {
        const long p0 = pbase + (long)t * TILE_P;

        // ---- stage: a -> transposed SoA, x -> padded float4 ----
        {
            const float4 av = ((const float4*)a)[p0 + tid];
            aT[0][tid] = av.x; aT[1][tid] = av.y;
            aT[2][tid] = av.z; aT[3][tid] = av.w;
            const float* xp = x + (p0 + tid) * 3;
            xS[tid] = make_float4(xp[0], xp[1], xp[2], 0.0f);
            __builtin_prefetch(&((const float4*)a)[p0 + TILE_P + tid], 0, 0);
        }
        __syncthreads();

        // ---- produce B fragments: each (p,s) phase exactly once ----
        {
            const int kb = wave * 32;
            #pragma unroll
            for (int j = 0; j < 16; ++j) {
                const float4 x0 = xS[kb + 2*j + 0];   // wave-wide broadcast
                const float4 x1 = xS[kb + 2*j + 1];
                const float ph0 = x0.x*xm0 + x0.y*xm1 + x0.z*xm2;
                const float ph1 = x1.x*xm0 + x1.y*xm1 + x1.z*xm2;
                breU[wave][lane][j] = pack_f16x2(__cosf(ph0), __cosf(ph1));
                bimU[wave][lane][j] = pack_f16x2(__sinf(ph0), __sinf(ph1));
            }
        }
        __syncthreads();

        // ---- consume: 4 K-steps, 8 WMMAs each ----
        #pragma unroll
        for (int kk = 0; kk < SUBSTEP; ++kk) {
            const int base = kk * 32;
            const int pr0  = base + 8 * half;        // run0: elements 0..7
            const int pr1  = pr0 + 16;               // run1: elements 8..15

            float accA[16], accB[16];
            #pragma unroll
            for (int q = 0; q < 16; ++q) { accA[q] = 0.0f; accB[q] = 0.0f; }

            #pragma unroll
            for (int ch = 0; ch < 4; ++ch) {
                const float4 u0 = *(const float4*)&aT[ch][pr0];
                const float4 u1 = *(const float4*)&aT[ch][pr0 + 4];
                const float4 w0 = *(const float4*)&aT[ch][pr1];
                const float4 w1 = *(const float4*)&aT[ch][pr1 + 4];
                const float uu[8] = {u0.x,u0.y,u0.z,u0.w,u1.x,u1.y,u1.z,u1.w};
                const float ww[8] = {w0.x,w0.y,w0.z,w0.w,w1.x,w1.y,w1.z,w1.w};
                const float sA = amA[ch], sB = amB[ch];
                #pragma unroll
                for (int q = 0; q < 8; ++q) {
                    accA[q]     += uu[q] * sA;
                    accA[8 + q] += ww[q] * sA;
                    accB[q]     += uu[q] * sB;
                    accB[8 + q] += ww[q] * sB;
                }
            }

            v16h afA, afB;
            #pragma unroll
            for (int e = 0; e < 16; ++e) {
                afA[e] = (_Float16)accA[e];
                afB[e] = (_Float16)accB[e];
            }

            #pragma unroll
            for (int st = 0; st < 2; ++st) {
                const unsigned* rre = &breU[kk][st*16 + n][4 * half];
                const unsigned* rim = &bimU[kk][st*16 + n][4 * half];
                const v16h bre = cat8(*(const v8h*)rre, *(const v8h*)(rre + 8));
                const v16h bim = cat8(*(const v8h*)rim, *(const v8h*)(rim + 8));

                accRe[st][0] = __builtin_amdgcn_wmma_f32_16x16x32_f16(
                    false, afA, false, bre, (short)0, accRe[st][0], false, false);
                accIm[st][0] = __builtin_amdgcn_wmma_f32_16x16x32_f16(
                    false, afA, false, bim, (short)0, accIm[st][0], false, false);
                accRe[st][1] = __builtin_amdgcn_wmma_f32_16x16x32_f16(
                    false, afB, false, bre, (short)0, accRe[st][1], false, false);
                accIm[st][1] = __builtin_amdgcn_wmma_f32_16x16x32_f16(
                    false, afB, false, bim, (short)0, accIm[st][1], false, false);
            }
        }
        __syncthreads();
    }

    // C/D layout: acc[r] -> M = r + 8*half, N = n. Scale already folded into A.
    #pragma unroll
    for (int st = 0; st < 2; ++st) {
        const int s = s0 + st*16 + n;
        #pragma unroll
        for (int r = 0; r < 8; ++r) {
            const int mA = m0a + r + 8*half;
            const int mB = m0b + r + 8*half;
            const long idxA = (((long)b * NMODES + mA) * NMODES + s) * 2;
            const long idxB = (((long)b * NMODES + mB) * NMODES + s) * 2;
            atomicAdd(&out[idxA + 0], accRe[st][0][r]);
            atomicAdd(&out[idxA + 1], accIm[st][0][r]);
            atomicAdd(&out[idxB + 0], accRe[st][1][r]);
            atomicAdd(&out[idxB + 1], accIm[st][1][r]);
        }
    }
}

extern "C" void kernel_launch(void* const* d_in, const int* in_sizes, int n_in,
                              void* d_out, int out_size, void* d_ws, size_t ws_size,
                              hipStream_t stream) {
    (void)in_sizes; (void)n_in; (void)out_size; (void)d_ws; (void)ws_size;
    const float* a       = (const float*)d_in[0];
    const float* x       = (const float*)d_in[1];
    const float* a_modes = (const float*)d_in[2];
    const float* x_modes = (const float*)d_in[3];
    float* out = (float*)d_out;

    SpectralTransform_zero<<<(OUT_FLOATS + 255) / 256, 256, 0, stream>>>(out);

    dim3 grid(SPLITK, NMODES / 32, NBATCH);   // 128 x 4 x 2 = 1024 workgroups
    SpectralTransform_61272003445097_kernel<<<grid, 128, 0, stream>>>(
        a, x, a_modes, x_modes, out);
}
